// EncoderBlock_20684562498083
// MI455X (gfx1250) — matrix operand order repaired and checked by reading, not compiled, not run on hardware
//
#include <hip/hip_runtime.h>

#define DIMV 1024
#define HEADSV 16
#define DKV 64
#define INNERV 3072
#define BATCHV 2
#define SEQV 2048
#define ROWSV (BATCHV * SEQV)   // 4096

typedef __attribute__((ext_vector_type(16))) __bf16 v16bf;
typedef __attribute__((ext_vector_type(8)))  float  v8f;

union FragU { uint4 q[2]; v16bf v; };

__device__ __forceinline__ unsigned short f2bf(float f) {
    unsigned int u = __float_as_uint(f);
    unsigned int r = (u + 0x7FFFu + ((u >> 16) & 1u)) >> 16;
    return (unsigned short)r;
}

// ---------------------------------------------------------------- converts
__global__ void f32_to_bf16_kernel(const float* __restrict__ in,
                                   unsigned short* __restrict__ out, int n) {
    int i = blockIdx.x * blockDim.x + threadIdx.x;
    if (i < n) out[i] = f2bf(in[i]);
}

// ---------------------------------------------------------------- rmsnorm
__global__ void rmsnorm_kernel(const float* __restrict__ x,
                               const float* __restrict__ w,
                               unsigned short* __restrict__ out) {
    __shared__ float red[256];
    int row = blockIdx.x;
    const float* xr = x + (size_t)row * DIMV;
    float s = 0.f;
    for (int i = threadIdx.x; i < DIMV; i += 256) { float v = xr[i]; s += v * v; }
    red[threadIdx.x] = s;
    __syncthreads();
    for (int off = 128; off > 0; off >>= 1) {
        if ((int)threadIdx.x < off) red[threadIdx.x] += red[threadIdx.x + off];
        __syncthreads();
    }
    float rs = rsqrtf(red[0] / (float)DIMV + 1e-5f);
    for (int i = threadIdx.x; i < DIMV; i += 256)
        out[(size_t)row * DIMV + i] = f2bf(w[i] * xr[i] * rs);
}

// ---------------------------------------------------------------- generic GEMM
// C(MxN,f32) = A(MxK,bf16) @ B(KxN,bf16) [+ residual]; block tile 128x128,
// 256 threads = 8 waves, wave tile 32x64. Double-buffered LDS (1 barrier/step).
#define LSTR 40
template <bool HASRES>
__global__ void gemm_kernel(const unsigned short* __restrict__ A,
                            const unsigned short* __restrict__ B,
                            const float* __restrict__ residual,
                            float* __restrict__ Cf,
                            int M, int N, int K) {
    __shared__ unsigned short Alds[2][128 * LSTR];
    __shared__ unsigned short Blds[2][128 * LSTR];
    int tid = threadIdx.x;
    int lane = tid & 31, wave = tid >> 5;
    int wm = (wave & 3) * 32, wn = (wave >> 2) * 64;
    int bm = blockIdx.y * 128, bn = blockIdx.x * 128;

    v8f acc[2][4];
    v8f vz = {0.f,0.f,0.f,0.f,0.f,0.f,0.f,0.f};
    for (int i = 0; i < 2; ++i) for (int j = 0; j < 4; ++j) acc[i][j] = vz;

    int arow = tid >> 1, acol = (tid & 1) * 16;
    int bkrow = tid >> 3, bnoff = (tid & 7) * 16;
    int koffA = (lane >= 16) ? 8 : 0;
    int koffB = (lane >= 16) ? 16 : 0;

    const unsigned short* aptr = A + (size_t)(bm + arow) * K + acol;
    const unsigned short* bptr = B + (size_t)bkrow * N + bn + bnoff;

    auto stage = [&](unsigned short* Ab, unsigned short* Bb, int kk) {
        const uint4* srcA = (const uint4*)(aptr + kk);
        uint4 a0 = srcA[0], a1 = srcA[1];
        *(uint4*)(&Ab[arow * LSTR + acol])     = a0;
        *(uint4*)(&Ab[arow * LSTR + acol + 8]) = a1;
        const unsigned short* srcB = bptr + (size_t)kk * N;
        unsigned short tmp[16];
        *(uint4*)(tmp)     = *(const uint4*)(srcB);
        *(uint4*)(tmp + 8) = *(const uint4*)(srcB + 8);
        #pragma unroll
        for (int i = 0; i < 16; ++i)
            Bb[(bnoff + i) * LSTR + bkrow] = tmp[i];
    };

    stage(Alds[0], Blds[0], 0);
    int buf = 0;
    for (int k0 = 0; k0 < K; k0 += 32) {
        __syncthreads();
        if (k0 + 32 < K) stage(Alds[buf ^ 1], Blds[buf ^ 1], k0 + 32);
        if (k0 + 64 < K) {
            __builtin_prefetch(aptr + k0 + 64, 0, 1);
            __builtin_prefetch(bptr + (size_t)(k0 + 64) * N, 0, 1);
        }
        const unsigned short* Ab = Alds[buf];
        const unsigned short* Bb = Blds[buf];

        v16bf af[2];
        #pragma unroll
        for (int mt = 0; mt < 2; ++mt) {
            const unsigned short* p = &Ab[(wm + mt * 16 + (lane & 15)) * LSTR];
            FragU u;
            u.q[0] = *(const uint4*)(p + koffA);
            u.q[1] = *(const uint4*)(p + 16 + koffA);
            af[mt] = u.v;
        }
        #pragma unroll
        for (int nt = 0; nt < 4; ++nt) {
            const unsigned short* p = &Bb[(wn + nt * 16 + (lane & 15)) * LSTR + koffA];
            FragU u;
            u.q[0] = *(const uint4*)(p);
            u.q[1] = *(const uint4*)(p + 16);
            v16bf bf = u.v;
            #pragma unroll
            for (int mt = 0; mt < 2; ++mt)
                acc[mt][nt] = __builtin_amdgcn_wmma_f32_16x16x32_bf16(
                    false, af[mt], false, bf, (short)0, acc[mt][nt], false, false);
        }
        buf ^= 1;
    }

    int rbase = (lane >= 16) ? 8 : 0;
    int col0 = bn + wn + (lane & 15);
    #pragma unroll
    for (int mt = 0; mt < 2; ++mt) {
        #pragma unroll
        for (int nt = 0; nt < 4; ++nt) {
            size_t idx = (size_t)(bm + wm + mt * 16 + rbase) * N + col0 + nt * 16;
            #pragma unroll
            for (int r = 0; r < 8; ++r) {
                float v = acc[mt][nt][r];
                if (HASRES) v += residual[idx];
                Cf[idx] = v;
                idx += N;
            }
        }
    }
}

// ---------------------------------------------------------------- fused fc + swiglu
// a = A@Wfc[:, 0:N), g = A@Wfc[:, N:2N); H = bf16(a * sigmoid(g)); N = INNER
// Block tile 128x64, 8 waves of 32x32 (8 accumulators/wave). Double-buffered.
__global__ void gemm_fc_swiglu_kernel(const unsigned short* __restrict__ A,
                                      const unsigned short* __restrict__ Wfc,
                                      unsigned short* __restrict__ H,
                                      int M, int N, int K) {
    __shared__ unsigned short Alds[2][128 * LSTR];
    __shared__ unsigned short Balds[2][64 * LSTR];
    __shared__ unsigned short Bglds[2][64 * LSTR];
    int tid = threadIdx.x;
    int lane = tid & 31, wave = tid >> 5;
    int wm = (wave & 3) * 32, wn = (wave >> 2) * 32;
    int bm = blockIdx.y * 128, bn = blockIdx.x * 64;
    int N2 = 2 * N;

    v8f acca[2][2], accg[2][2];
    v8f vz = {0.f,0.f,0.f,0.f,0.f,0.f,0.f,0.f};
    for (int i = 0; i < 2; ++i) for (int j = 0; j < 2; ++j) { acca[i][j] = vz; accg[i][j] = vz; }

    int arow = tid >> 1, acol = (tid & 1) * 16;
    int bkrow = tid >> 3, bnoff = (tid & 7) * 8;   // 32 k-rows x 64 n, 8 elems/thread
    int koffA = (lane >= 16) ? 8 : 0;
    int koffB = (lane >= 16) ? 16 : 0;

    const unsigned short* aptr = A + (size_t)(bm + arow) * K + acol;
    const unsigned short* bptr = Wfc + (size_t)bkrow * N2 + bn + bnoff;

    auto stage = [&](unsigned short* Ab, unsigned short* Bab, unsigned short* Bgb, int kk) {
        const uint4* srcA = (const uint4*)(aptr + kk);
        uint4 a0 = srcA[0], a1 = srcA[1];
        *(uint4*)(&Ab[arow * LSTR + acol])     = a0;
        *(uint4*)(&Ab[arow * LSTR + acol + 8]) = a1;
        const unsigned short* sa = bptr + (size_t)kk * N2;
        const unsigned short* sg = sa + N;
        unsigned short ta[8], tg[8];
        *(uint4*)(ta) = *(const uint4*)(sa);
        *(uint4*)(tg) = *(const uint4*)(sg);
        #pragma unroll
        for (int i = 0; i < 8; ++i) {
            Bab[(bnoff + i) * LSTR + bkrow] = ta[i];
            Bgb[(bnoff + i) * LSTR + bkrow] = tg[i];
        }
    };

    stage(Alds[0], Balds[0], Bglds[0], 0);
    int buf = 0;
    for (int k0 = 0; k0 < K; k0 += 32) {
        __syncthreads();
        if (k0 + 32 < K) stage(Alds[buf ^ 1], Balds[buf ^ 1], Bglds[buf ^ 1], k0 + 32);
        if (k0 + 64 < K) {
            __builtin_prefetch(aptr + k0 + 64, 0, 1);
            __builtin_prefetch(bptr + (size_t)(k0 + 64) * N2, 0, 1);
            __builtin_prefetch(bptr + (size_t)(k0 + 64) * N2 + N, 0, 1);
        }
        const unsigned short* Ab  = Alds[buf];
        const unsigned short* Bab = Balds[buf];
        const unsigned short* Bgb = Bglds[buf];

        v16bf af[2];
        #pragma unroll
        for (int mt = 0; mt < 2; ++mt) {
            const unsigned short* p = &Ab[(wm + mt * 16 + (lane & 15)) * LSTR];
            FragU u;
            u.q[0] = *(const uint4*)(p + koffA);
            u.q[1] = *(const uint4*)(p + 16 + koffA);
            af[mt] = u.v;
        }
        #pragma unroll
        for (int nt = 0; nt < 2; ++nt) {
            int coff = (wn + nt * 16 + (lane & 15)) * LSTR + koffA;
            FragU ua, ug;
            ua.q[0] = *(const uint4*)(&Bab[coff]);
            ua.q[1] = *(const uint4*)(&Bab[coff + 16]);
            ug.q[0] = *(const uint4*)(&Bgb[coff]);
            ug.q[1] = *(const uint4*)(&Bgb[coff + 16]);
            #pragma unroll
            for (int mt = 0; mt < 2; ++mt) {
                acca[mt][nt] = __builtin_amdgcn_wmma_f32_16x16x32_bf16(
                    false, af[mt], false, ua.v, (short)0, acca[mt][nt], false, false);
                accg[mt][nt] = __builtin_amdgcn_wmma_f32_16x16x32_bf16(
                    false, af[mt], false, ug.v, (short)0, accg[mt][nt], false, false);
            }
        }
        buf ^= 1;
    }

    int rbase = (lane >= 16) ? 8 : 0;
    int col0 = bn + wn + (lane & 15);
    #pragma unroll
    for (int mt = 0; mt < 2; ++mt) {
        #pragma unroll
        for (int nt = 0; nt < 2; ++nt) {
            size_t idx = (size_t)(bm + wm + mt * 16 + rbase) * N + col0 + nt * 16;
            #pragma unroll
            for (int r = 0; r < 8; ++r) {
                float a = acca[mt][nt][r];
                float g = accg[mt][nt][r];
                float h = a * (1.f / (1.f + __expf(-g)));
                H[idx] = f2bf(h);
                idx += N;
            }
        }
    }
}

// ---------------------------------------------------------------- rope + split
__global__ void rope_split_kernel(const float* __restrict__ qkv,
                                  const float* __restrict__ rope,
                                  unsigned short* __restrict__ qh,
                                  unsigned short* __restrict__ kh,
                                  unsigned short* __restrict__ vh) {
    int idx = blockIdx.x * blockDim.x + threadIdx.x;   // ROWS * 512 pairs
    if (idx >= ROWSV * (DIMV / 2)) return;
    int row = idx >> 9;
    int p   = idx & 511;
    int t   = row & (SEQV - 1);
    int b   = row >> 11;
    float c = rope[((size_t)t * 512 + p) * 2 + 0];
    float s = rope[((size_t)t * 512 + p) * 2 + 1];
    int d0 = p * 2;
    const float* base = qkv + (size_t)row * 3 * DIMV;
    float qr = base[d0],            qi = base[d0 + 1];
    float kr = base[DIMV + d0],     ki = base[DIMV + d0 + 1];
    float v0 = base[2 * DIMV + d0], v1 = base[2 * DIMV + d0 + 1];
    float qor = qr * c - qi * s, qoi = qr * s + qi * c;
    float kor = kr * c - ki * s, koi = kr * s + ki * c;
    int h = d0 >> 6, dk = d0 & 63;
    size_t o = ((size_t)(b * HEADSV + h) * SEQV + t) * DKV + dk;
    qh[o] = f2bf(qor); qh[o + 1] = f2bf(qoi);
    kh[o] = f2bf(kor); kh[o + 1] = f2bf(koi);
    vh[o] = f2bf(v0);  vh[o + 1] = f2bf(v1);
}

// ---------------------------------------------------------------- flash attention
// grid (SEQ/64, B*HEADS), block 128 (4 waves, 16 q-rows each). K/V double-buffered.
#define KVSTR 72
__global__ void flash_attn_kernel(const unsigned short* __restrict__ qh,
                                  const unsigned short* __restrict__ kh,
                                  const unsigned short* __restrict__ vh,
                                  unsigned short* __restrict__ attnout) {
    __shared__ unsigned short Klds[2][64 * KVSTR];
    __shared__ unsigned short Vtlds[2][64 * KVSTR];
    __shared__ unsigned short Plds[64 * KVSTR];
    int qblk = blockIdx.x;
    int bh   = blockIdx.y;
    int b = bh / HEADSV, h = bh % HEADSV;
    int tid = threadIdx.x, lane = tid & 31, wave = tid >> 5;
    int rbase = (lane >= 16) ? 8 : 0;
    int koffA = (lane >= 16) ? 8 : 0;
    int koffB = (lane >= 16) ? 16 : 0;
    size_t head_off = (size_t)bh * SEQV * DKV;

    // Q fragments (held in registers for whole kv loop)
    v16bf qf[2];
    {
        int qrow = qblk * 64 + wave * 16 + (lane & 15);
        const unsigned short* qp = qh + head_off + (size_t)qrow * DKV;
        #pragma unroll
        for (int ks = 0; ks < 2; ++ks) {
            FragU u;
            u.q[0] = *(const uint4*)(qp + ks * 32 + koffA);
            u.q[1] = *(const uint4*)(qp + ks * 32 + 16 + koffA);
            qf[ks] = u.v;
        }
    }

    v8f acc_o[4];
    v8f vz = {0.f,0.f,0.f,0.f,0.f,0.f,0.f,0.f};
    for (int i = 0; i < 4; ++i) acc_o[i] = vz;
    float mrow[8], lrow[8];
    #pragma unroll
    for (int r = 0; r < 8; ++r) { mrow[r] = -3.0e38f; lrow[r] = 0.f; }

    int key = tid >> 1, dkoff = (tid & 1) * 32;
    const unsigned short* kbase = kh + head_off + (size_t)key * DKV + dkoff;
    const unsigned short* vbase = vh + head_off + (size_t)key * DKV + dkoff;

    auto stage_kv = [&](unsigned short* Kb, unsigned short* Vb, int kv) {
        const uint4* src = (const uint4*)(kbase + (size_t)kv * DKV);
        #pragma unroll
        for (int i = 0; i < 4; ++i)
            *(uint4*)(&Kb[key * KVSTR + dkoff + i * 8]) = src[i];
        const uint4* vsrc = (const uint4*)(vbase + (size_t)kv * DKV);
        unsigned short tmp[32];
        #pragma unroll
        for (int i = 0; i < 4; ++i) *(uint4*)(tmp + i * 8) = vsrc[i];
        #pragma unroll
        for (int i = 0; i < 32; ++i)
            Vb[(dkoff + i) * KVSTR + key] = tmp[i];
    };

    stage_kv(Klds[0], Vtlds[0], 0);
    int buf = 0;
    for (int kv0 = 0; kv0 < SEQV; kv0 += 64) {
        __syncthreads();
        if (kv0 + 64 < SEQV) stage_kv(Klds[buf ^ 1], Vtlds[buf ^ 1], kv0 + 64);
        if (kv0 + 128 < SEQV) {
            __builtin_prefetch(kbase + (size_t)(kv0 + 128) * DKV, 0, 1);
            __builtin_prefetch(vbase + (size_t)(kv0 + 128) * DKV, 0, 1);
        }
        const unsigned short* Kb = Klds[buf];
        const unsigned short* Vb = Vtlds[buf];

        // S = (Q K^T) * 0.125
        v8f s_acc[4];
        #pragma unroll
        for (int nt = 0; nt < 4; ++nt) {
            s_acc[nt] = vz;
            #pragma unroll
            for (int ks = 0; ks < 2; ++ks) {
                const unsigned short* p = &Kb[(nt * 16 + (lane & 15)) * KVSTR + ks * 32 + koffA];
                FragU u;
                u.q[0] = *(const uint4*)(p);
                u.q[1] = *(const uint4*)(p + 16);
                s_acc[nt] = __builtin_amdgcn_wmma_f32_16x16x32_bf16(
                    false, qf[ks], false, u.v, (short)0, s_acc[nt], false, false);
            }
        }
        #pragma unroll
        for (int nt = 0; nt < 4; ++nt)
            #pragma unroll
            for (int r = 0; r < 8; ++r) s_acc[nt][r] *= 0.125f;

        // running softmax update
        #pragma unroll
        for (int r = 0; r < 8; ++r) {
            float mx = s_acc[0][r];
            mx = fmaxf(mx, s_acc[1][r]);
            mx = fmaxf(mx, s_acc[2][r]);
            mx = fmaxf(mx, s_acc[3][r]);
            mx = fmaxf(mx, __shfl_xor(mx, 1, 32));
            mx = fmaxf(mx, __shfl_xor(mx, 2, 32));
            mx = fmaxf(mx, __shfl_xor(mx, 4, 32));
            mx = fmaxf(mx, __shfl_xor(mx, 8, 32));
            float mnew  = fmaxf(mrow[r], mx);
            float alpha = __expf(mrow[r] - mnew);
            mrow[r] = mnew;
            lrow[r] *= alpha;
            #pragma unroll
            for (int nt = 0; nt < 4; ++nt) acc_o[nt][r] *= alpha;
            float ps = 0.f;
            #pragma unroll
            for (int nt = 0; nt < 4; ++nt) {
                float pv = __expf(s_acc[nt][r] - mnew);
                ps += pv;
                Plds[(wave * 16 + rbase + r) * KVSTR + nt * 16 + (lane & 15)] = f2bf(pv);
            }
            ps += __shfl_xor(ps, 1, 32);
            ps += __shfl_xor(ps, 2, 32);
            ps += __shfl_xor(ps, 4, 32);
            ps += __shfl_xor(ps, 8, 32);
            lrow[r] += ps;
        }

        // O += P @ V   (P restaged via LDS -> A-fragment layout; same-wave DS is in-order)
        #pragma unroll
        for (int nt = 0; nt < 4; ++nt) {
            #pragma unroll
            for (int ks = 0; ks < 2; ++ks) {
                const unsigned short* pp = &Plds[(wave * 16 + (lane & 15)) * KVSTR + ks * 32];
                FragU ua;
                ua.q[0] = *(const uint4*)(pp + koffA);
                ua.q[1] = *(const uint4*)(pp + 16 + koffA);
                const unsigned short* vp = &Vb[(nt * 16 + (lane & 15)) * KVSTR + ks * 32 + koffA];
                FragU ub;
                ub.q[0] = *(const uint4*)(vp);
                ub.q[1] = *(const uint4*)(vp + 16);
                acc_o[nt] = __builtin_amdgcn_wmma_f32_16x16x32_bf16(
                    false, ua.v, false, ub.v, (short)0, acc_o[nt], false, false);
            }
        }
        buf ^= 1;
    }

    // normalize + write bf16 attn output in (B*T, H*DK) layout
    {
        int col0 = h * DKV + (lane & 15);
        size_t idx0 = (size_t)(b * SEQV + qblk * 64 + wave * 16 + rbase) * DIMV + col0;
        #pragma unroll
        for (int r = 0; r < 8; ++r) {
            float inv = 1.f / lrow[r];
            size_t idx = idx0 + (size_t)r * DIMV;
            #pragma unroll
            for (int nt = 0; nt < 4; ++nt)
                attnout[idx + nt * 16] = f2bf(acc_o[nt][r] * inv);
        }
    }
}

// ---------------------------------------------------------------- host
extern "C" void kernel_launch(void* const* d_in, const int* in_sizes, int n_in,
                              void* d_out, int out_size, void* d_ws, size_t ws_size,
                              hipStream_t stream) {
    const float* x    = (const float*)d_in[0];
    const float* rope = (const float*)d_in[1];
    const float* wn1  = (const float*)d_in[2];
    const float* wqkv = (const float*)d_in[3];
    const float* wap  = (const float*)d_in[4];
    const float* wn2  = (const float*)d_in[5];
    const float* wfc  = (const float*)d_in[6];
    const float* wffn = (const float*)d_in[7];
    float* out = (float*)d_out;

    char* ws = (char*)d_ws;
    auto alloc = [&](size_t bytes) -> void* {
        void* p = (void*)ws;
        ws += (bytes + 255) & ~(size_t)255;
        return p;
    };
    unsigned short* xn      = (unsigned short*)alloc((size_t)ROWSV * DIMV * 2);
    float*          qkv     = (float*)         alloc((size_t)ROWSV * 3 * DIMV * 4);
    unsigned short* qh      = (unsigned short*)alloc((size_t)ROWSV * DIMV * 2);
    unsigned short* kh      = (unsigned short*)alloc((size_t)ROWSV * DIMV * 2);
    unsigned short* vh      = (unsigned short*)alloc((size_t)ROWSV * DIMV * 2);
    unsigned short* attnout = (unsigned short*)alloc((size_t)ROWSV * DIMV * 2);
    float*          x1      = (float*)         alloc((size_t)ROWSV * DIMV * 4);
    unsigned short* xn2     = (unsigned short*)alloc((size_t)ROWSV * DIMV * 2);
    unsigned short* hbuf    = (unsigned short*)alloc((size_t)ROWSV * INNERV * 2);
    unsigned short* wqkv_b  = (unsigned short*)alloc((size_t)DIMV * 3 * DIMV * 2);
    unsigned short* wap_b   = (unsigned short*)alloc((size_t)DIMV * DIMV * 2);
    unsigned short* wfc_b   = (unsigned short*)alloc((size_t)DIMV * 2 * INNERV * 2);
    unsigned short* wffn_b  = (unsigned short*)alloc((size_t)INNERV * DIMV * 2);

    // weight conversion (fp32 -> bf16)
    {
        int n;
        n = DIMV * 3 * DIMV;
        f32_to_bf16_kernel<<<(n + 255) / 256, 256, 0, stream>>>(wqkv, wqkv_b, n);
        n = DIMV * DIMV;
        f32_to_bf16_kernel<<<(n + 255) / 256, 256, 0, stream>>>(wap, wap_b, n);
        n = DIMV * 2 * INNERV;
        f32_to_bf16_kernel<<<(n + 255) / 256, 256, 0, stream>>>(wfc, wfc_b, n);
        n = INNERV * DIMV;
        f32_to_bf16_kernel<<<(n + 255) / 256, 256, 0, stream>>>(wffn, wffn_b, n);
    }

    // 1) xn = rmsnorm(x, w_norm1)
    rmsnorm_kernel<<<ROWSV, 256, 0, stream>>>(x, wn1, xn);
    // 2) qkv = xn @ w_qkv
    gemm_kernel<false><<<dim3(3 * DIMV / 128, ROWSV / 128), 256, 0, stream>>>(
        xn, wqkv_b, nullptr, qkv, ROWSV, 3 * DIMV, DIMV);
    // 3) rope + split to head-major bf16
    {
        int n = ROWSV * (DIMV / 2);
        rope_split_kernel<<<(n + 255) / 256, 256, 0, stream>>>(qkv, rope, qh, kh, vh);
    }
    // 4) attention
    flash_attn_kernel<<<dim3(SEQV / 64, BATCHV * HEADSV), 128, 0, stream>>>(
        qh, kh, vh, attnout);
    // 5) x1 = x + attnout @ w_attnproj
    gemm_kernel<true><<<dim3(DIMV / 128, ROWSV / 128), 256, 0, stream>>>(
        attnout, wap_b, x, x1, ROWSV, DIMV, DIMV);
    // 6) xn2 = rmsnorm(x1, w_norm2)
    rmsnorm_kernel<<<ROWSV, 256, 0, stream>>>(x1, wn2, xn2);
    // 7) h = swiglu(xn2 @ w_fc)  (block 128x64, dual a/g accumulation)
    gemm_fc_swiglu_kernel<<<dim3(INNERV / 64, ROWSV / 128), 256, 0, stream>>>(
        xn2, wfc_b, hbuf, ROWSV, INNERV, DIMV);
    // 8) out = x1 + h @ w_ffnproj
    gemm_kernel<true><<<dim3(DIMV / 128, ROWSV / 128), 256, 0, stream>>>(
        hbuf, wffn_b, x1, out, ROWSV, DIMV, INNERV);
    (void)in_sizes; (void)n_in; (void)out_size; (void)ws_size;
}
